// LSTMVAE_64089501991277
// MI455X (gfx1250) — matrix operand order repaired and checked by reading, not compiled, not run on hardware
//
#include <hip/hip_runtime.h>
#include <hip/hip_bf16.h>
#include <stdint.h>
#include <stddef.h>

#define Bdim 32
#define Tdim 128
#define Edim 512
#define Hdim 1024
#define Sdim 64
#define Cdim 192
#define Vdim 32000
#define REC_BLOCKS 16

typedef __attribute__((__ext_vector_type__(16))) __bf16 v16bf;
typedef __attribute__((__ext_vector_type__(8)))  __bf16 v8bf;
typedef __attribute__((__ext_vector_type__(8)))  float  v8f;

__device__ __forceinline__ float sigf(float x) { return 1.0f / (1.0f + __expf(-x)); }

__device__ __forceinline__ v8f wmma_bf16(v16bf a, v16bf b, v8f c) {
  return __builtin_amdgcn_wmma_f32_16x16x32_bf16(false, a, false, b, (short)0, c, false, false);
}

// A fragment (bf16 16x32, row-major [M,K]). Arow already includes
//   + m*K + 16*(lane>>4). One contiguous 16-elem run at k0.
__device__ __forceinline__ v16bf load_a(const __bf16* Arow, int k0) {
  union { v16bf v; v8bf h[2]; } u;
  u.h[0] = *(const v8bf*)(Arow + k0);
  u.h[1] = *(const v8bf*)(Arow + k0 + 8);
  return u.v;
}

// CDNA5 async global->LDS copy (per-lane B128), tracked by ASYNCcnt.
__device__ __forceinline__ void async_ld_b128(unsigned lds_addr, const __bf16* gaddr) {
  asm volatile("global_load_async_to_lds_b128 %0, %1, off"
               :: "v"(lds_addr), "v"(gaddr) : "memory");
}
__device__ __forceinline__ void wait_async0() {
  asm volatile("s_wait_asynccnt 0x0" ::: "memory");
}
__device__ __forceinline__ void wait_async4() {
  asm volatile("s_wait_asynccnt 0x4" ::: "memory");
}

// ---------------------------------------------------------------------------
// f32 -> bf16 flat convert
__global__ void cvt_kernel(const float* __restrict__ s, __bf16* __restrict__ d, size_t n) {
  size_t i = (size_t)blockIdx.x * blockDim.x + threadIdx.x;
  size_t stride = (size_t)gridDim.x * blockDim.x;
  for (; i < n; i += stride) d[i] = (__bf16)s[i];
}

__global__ void add2_kernel(const float* __restrict__ a, const float* __restrict__ b,
                            float* __restrict__ d, int n) {
  int i = blockIdx.x * blockDim.x + threadIdx.x;
  if (i < n) d[i] = a[i] + b[i];
}

// ---------------------------------------------------------------------------
// Embedding gather + LayerNorm. Writes bf16 xt[t][b][e] (time-major) and
// shifted decoder input decin[t+1][b][e]. Block = one (b,t) token, 128 thr.
__global__ void embed_ln_kernel(const int* __restrict__ x, const float* __restrict__ emb,
                                const float* __restrict__ g, const float* __restrict__ be,
                                __bf16* __restrict__ xt, __bf16* __restrict__ decin) {
  const int bt = blockIdx.x;           // b*T + t  (x is [B,T])
  const int bb = bt >> 7;
  const int tt = bt & (Tdim - 1);
  const float* row = emb + (size_t)x[bt] * Edim;
  __shared__ float s1[128], s2[128];
  float s = 0.f, q = 0.f;
  for (int e = threadIdx.x; e < Edim; e += 128) { float v = row[e]; s += v; q += v * v; }
  s1[threadIdx.x] = s; s2[threadIdx.x] = q;
  __syncthreads();
  for (int off = 64; off > 0; off >>= 1) {
    if ((int)threadIdx.x < off) { s1[threadIdx.x] += s1[threadIdx.x + off]; s2[threadIdx.x] += s2[threadIdx.x + off]; }
    __syncthreads();
  }
  const float mean = s1[0] * (1.0f / Edim);
  const float var  = s2[0] * (1.0f / Edim) - mean * mean;
  const float rstd = rsqrtf(var + 1e-5f);
  for (int e = threadIdx.x; e < Edim; e += 128) {
    float v = (row[e] - mean) * rstd * g[e] + be[e];
    xt[((size_t)tt * Bdim + bb) * Edim + e] = (__bf16)v;
    if (tt + 1 < Tdim) decin[((size_t)(tt + 1) * Bdim + bb) * Edim + e] = (__bf16)v;
  }
}

// decin[0][b][:] = emb[sos]  (raw embedding, NOT layernormed)
__global__ void sos_kernel(const float* __restrict__ emb, const int* __restrict__ sos,
                           __bf16* __restrict__ decin) {
  int i = blockIdx.x * blockDim.x + threadIdx.x;
  if (i >= Bdim * Edim) return;
  int e = i % Edim;
  decin[i] = (__bf16)emb[(size_t)sos[0] * Edim + e];
}

// ---------------------------------------------------------------------------
// Small GEMM (M multiple of 16): C[M,N] = A(bf16 [M,K]) @ W(bf16 [N,K])^T + b.
// One wave: 16x64 tile, fragments straight from global.
__global__ void gemm_kernel(const __bf16* __restrict__ A, const __bf16* __restrict__ W,
                            const float* __restrict__ bias, float* __restrict__ C,
                            int M, int N, int K) {
  const int lane = threadIdx.x & 31;
  const int gw = (blockIdx.x * blockDim.x + threadIdx.x) >> 5;
  const int ntn = N >> 6;
  if (gw >= (M >> 4) * ntn) return;            // wave-uniform exit (EXEC stays full)
  const int m0 = (gw / ntn) << 4;
  const int n0 = (gw % ntn) << 6;
  const __bf16* Arow = A + (size_t)(m0 + (lane & 15)) * K + 16 * (lane >> 4);
  const int ko = 16 * (lane >> 4);
  const __bf16* W0 = W + (size_t)(n0 +  0 + (lane & 15)) * K + ko;
  const __bf16* W1 = W + (size_t)(n0 + 16 + (lane & 15)) * K + ko;
  const __bf16* W2 = W + (size_t)(n0 + 32 + (lane & 15)) * K + ko;
  const __bf16* W3 = W + (size_t)(n0 + 48 + (lane & 15)) * K + ko;
  v8f a0 = {}, a1 = {}, a2 = {}, a3 = {};
  for (int k0 = 0; k0 < K; k0 += 32) {
    v16bf av = load_a(Arow, k0);
    a0 = wmma_bf16(av, *(const v16bf*)(W0 + k0), a0);
    a1 = wmma_bf16(av, *(const v16bf*)(W1 + k0), a1);
    a2 = wmma_bf16(av, *(const v16bf*)(W2 + k0), a2);
    a3 = wmma_bf16(av, *(const v16bf*)(W3 + k0), a3);
  }
  const int r0 = m0 + 8 * (lane >> 4);
  const int c0 = n0 + (lane & 15);
  #pragma unroll
  for (int j = 0; j < 8; ++j) {
    float* Cr = C + (size_t)(r0 + j) * N;
    Cr[c0     ] = a0[j] + (bias ? bias[c0     ] : 0.f);
    Cr[c0 + 16] = a1[j] + (bias ? bias[c0 + 16] : 0.f);
    Cr[c0 + 32] = a2[j] + (bias ? bias[c0 + 32] : 0.f);
    Cr[c0 + 48] = a3[j] + (bias ? bias[c0 + 48] : 0.f);
  }
}

// ---------------------------------------------------------------------------
// Blocked GEMM with async-LDS W staging. WG tile 128(M) x 128(N); 8 waves,
// each wave 16 rows x 128 cols (8 accumulators). W tile (128 rows x 64 K) is
// double-buffered in LDS via global_load_async_to_lds_b128 (ASYNCcnt), so
// each W element is read from L2 once per 128 output rows instead of per 16.
// permute: 0 -> C row = r ; 1 -> C row = (r%B)*T + r/B  (vocab projection).
__global__ void gemm_lds_kernel(const __bf16* __restrict__ A, const __bf16* __restrict__ W,
                                const float* __restrict__ bias, float* __restrict__ C,
                                int M, int N, int K, int permute) {
  __shared__ __bf16 Wtile[2][128 * 64];        // 2 x 16KB
  const int lane = threadIdx.x & 31;
  const int wv = threadIdx.x >> 5;
  const int tid = threadIdx.x;
  const int ntn = N >> 7;                      // N/128
  const int mtile = blockIdx.x / ntn;
  const int ntile = blockIdx.x % ntn;
  const int m0 = (mtile << 7) + (wv << 4);     // wave's 16 rows
  const int n0 = ntile << 7;
  const int nk = K >> 6;                       // 64-wide K tiles

  const __bf16* Arow = A + (size_t)(m0 + (lane & 15)) * K + 16 * (lane >> 4);
  // per-thread staging addresses: 32 contiguous bf16 (4 x B128 per tile)
  const unsigned lds0 = (unsigned)(uintptr_t)(&Wtile[0][0]);
  const unsigned lds1 = (unsigned)(uintptr_t)(&Wtile[1][0]);
  const int sidx = tid * 32;                   // flat elem index in 128x64 tile
  const int sn = sidx >> 6;                    // staged row (n within tile)
  const int sk = sidx & 63;                    // staged k-offset
  const __bf16* Sg = W + (size_t)(n0 + sn) * K + sk;

  v8f acc[8];
  #pragma unroll
  for (int tb = 0; tb < 8; ++tb) acc[tb] = (v8f){};

  // prologue: stage tile 0 into buffer 0
  #pragma unroll
  for (int q = 0; q < 4; ++q)
    async_ld_b128(lds0 + (unsigned)(sidx + q * 8) * 2, Sg + q * 8);

  for (int kb = 0; kb < nk; ++kb) {
    const int cur = kb & 1;
    if (kb + 1 < nk) {                         // stage next tile into other buf
      const unsigned ldsn = (cur ? lds0 : lds1);
      const __bf16* Sgn = Sg + (size_t)(kb + 1) * 64;
      #pragma unroll
      for (int q = 0; q < 4; ++q)
        async_ld_b128(ldsn + (unsigned)(sidx + q * 8) * 2, Sgn + q * 8);
      wait_async4();                           // in-order: current tile landed
    } else {
      wait_async0();
    }
    __syncthreads();                           // tile visible to all waves
    const __bf16* Wl = &Wtile[cur][0] + (size_t)(lane & 15) * 64 + 16 * (lane >> 4);
    #pragma unroll
    for (int kc = 0; kc < 64; kc += 32) {
      v16bf av = load_a(Arow, kb * 64 + kc);
      #pragma unroll
      for (int tb = 0; tb < 8; ++tb) {
        v16bf wf = *(const v16bf*)(Wl + (size_t)tb * (16 * 64) + kc);
        acc[tb] = wmma_bf16(av, wf, acc[tb]);
      }
    }
    __syncthreads();                           // done reading before re-stage
  }

  const int r0 = m0 + 8 * (lane >> 4);
  #pragma unroll
  for (int j = 0; j < 8; ++j) {
    const int r = r0 + j;
    const int orow = permute ? ((r & (Bdim - 1)) * Tdim + (r >> 5)) : r;
    float* Cr = C + (size_t)orow * N;
    #pragma unroll
    for (int tb = 0; tb < 8; ++tb) {
      const int c = n0 + tb * 16 + (lane & 15);
      Cr[c] = acc[tb][j] + bias[c];
    }
  }
}

// ---------------------------------------------------------------------------
// Grid-wide barrier: monotonic counter, REC_BLOCKS co-resident workgroups.
__device__ __forceinline__ void grid_barrier(unsigned* bar, unsigned target) {
  __threadfence();
  __syncthreads();
  if (threadIdx.x == 0) {
    atomicAdd(bar, 1u);
    while (atomicAdd(bar, 0u) < target) {
#if defined(__HIP_DEVICE_COMPILE__)
      __builtin_amdgcn_s_sleep(1);
#endif
    }
    __threadfence();
  }
  __syncthreads();
}

// Persistent LSTM recurrence (one layer per launch).
// gi  : f32 [T, B, 4H]  (x@Wih.T + bih + bhh precomputed)
// whh : bf16 [4H, H]; hbuf: bf16 ping-pong [2][B][H]
// Each wave owns a 16(batch) x 16(h-col) tile; c state lives in registers.
__global__ void lstm_rec_kernel(const float* __restrict__ gi, const __bf16* __restrict__ whh,
                                __bf16* __restrict__ hbuf, __bf16* __restrict__ ys,
                                float* __restrict__ hfinal, unsigned* __restrict__ bar) {
  const int lane = threadIdx.x & 31;
  const int wv = threadIdx.x >> 5;
  const int m0 = (wv & 1) * 16;
  const int col0 = blockIdx.x * 64 + (wv >> 1) * 16;
  const int r0 = m0 + 8 * (lane >> 4);
  const int cc = col0 + (lane & 15);
  const int ko = 16 * (lane >> 4);
  const __bf16* Wl = whh + (size_t)(col0 + (lane & 15)) * Hdim + ko;  // gate i rows
  float creg[8];
  #pragma unroll
  for (int j = 0; j < 8; ++j) {
    creg[j] = 0.f;
    hbuf[(size_t)(r0 + j) * Hdim + cc] = (__bf16)0.f;    // h0 = 0 in buffer 0
  }
  grid_barrier(bar, (unsigned)REC_BLOCKS);
  for (int t = 0; t < Tdim; ++t) {
    const __bf16* hin = hbuf + (size_t)(t & 1) * Bdim * Hdim;
    const __bf16* Arow = hin + (size_t)(m0 + (lane & 15)) * Hdim + 16 * (lane >> 4);
    v8f ai = {}, af = {}, ag = {}, ao = {};
    for (int k0 = 0; k0 < Hdim; k0 += 32) {
      v16bf av = load_a(Arow, k0);
      ai = wmma_bf16(av, *(const v16bf*)(Wl + k0), ai);
      af = wmma_bf16(av, *(const v16bf*)(Wl + ((size_t)1 << 20) + k0), af);
      ag = wmma_bf16(av, *(const v16bf*)(Wl + ((size_t)2 << 20) + k0), ag);
      ao = wmma_bf16(av, *(const v16bf*)(Wl + ((size_t)3 << 20) + k0), ao);
    }
    const float* git = gi + (size_t)t * Bdim * (4 * Hdim);
    __bf16* hout = hbuf + (size_t)((t + 1) & 1) * Bdim * Hdim;
    #pragma unroll
    for (int j = 0; j < 8; ++j) {
      const int r = r0 + j;
      const float* gr = git + (size_t)r * (4 * Hdim);
      float pi = ai[j] + gr[cc];
      float pf = af[j] + gr[Hdim + cc];
      float pg = ag[j] + gr[2 * Hdim + cc];
      float po = ao[j] + gr[3 * Hdim + cc];
      float c = sigf(pf) * creg[j] + sigf(pi) * tanhf(pg);
      creg[j] = c;
      float h = sigf(po) * tanhf(c);
      hout[(size_t)r * Hdim + cc] = (__bf16)h;
      if (ys) ys[((size_t)t * Bdim + r) * Hdim + cc] = (__bf16)h;
      if (t == Tdim - 1) hfinal[(size_t)r * Hdim + cc] = h;
    }
    grid_barrier(bar, (unsigned)REC_BLOCKS * (t + 2));
  }
}

// Persistent GRU recurrence. gi: f32 [T,B,3H] (x@Wih.T + bih).
// whh bf16 [3H,H]; bhh f32 [3H]; zinit f32 [B,H] = initial hidden.
__global__ void gru_rec_kernel(const float* __restrict__ gi, const __bf16* __restrict__ whh,
                               const float* __restrict__ bhh, const float* __restrict__ zinit,
                               __bf16* __restrict__ hbuf, __bf16* __restrict__ ys,
                               unsigned* __restrict__ bar) {
  const int lane = threadIdx.x & 31;
  const int wv = threadIdx.x >> 5;
  const int m0 = (wv & 1) * 16;
  const int col0 = blockIdx.x * 64 + (wv >> 1) * 16;
  const int r0 = m0 + 8 * (lane >> 4);
  const int cc = col0 + (lane & 15);
  const int ko = 16 * (lane >> 4);
  const __bf16* Wl = whh + (size_t)(col0 + (lane & 15)) * Hdim + ko;
  float hreg[8];
  #pragma unroll
  for (int j = 0; j < 8; ++j) {
    float h = zinit[(size_t)(r0 + j) * Hdim + cc];
    hreg[j] = h;
    hbuf[(size_t)(r0 + j) * Hdim + cc] = (__bf16)h;
  }
  grid_barrier(bar, (unsigned)REC_BLOCKS);
  for (int t = 0; t < Tdim; ++t) {
    const __bf16* hin = hbuf + (size_t)(t & 1) * Bdim * Hdim;
    const __bf16* Arow = hin + (size_t)(m0 + (lane & 15)) * Hdim + 16 * (lane >> 4);
    v8f ar = {}, az = {}, an = {};
    for (int k0 = 0; k0 < Hdim; k0 += 32) {
      v16bf av = load_a(Arow, k0);
      ar = wmma_bf16(av, *(const v16bf*)(Wl + k0), ar);
      az = wmma_bf16(av, *(const v16bf*)(Wl + ((size_t)1 << 20) + k0), az);
      an = wmma_bf16(av, *(const v16bf*)(Wl + ((size_t)2 << 20) + k0), an);
    }
    const float* git = gi + (size_t)t * Bdim * (3 * Hdim);
    __bf16* hout = hbuf + (size_t)((t + 1) & 1) * Bdim * Hdim;
    #pragma unroll
    for (int j = 0; j < 8; ++j) {
      const int r = r0 + j;
      const float* gr = git + (size_t)r * (3 * Hdim);
      float ghr = ar[j] + bhh[cc];
      float ghz = az[j] + bhh[Hdim + cc];
      float ghn = an[j] + bhh[2 * Hdim + cc];
      float rg = sigf(gr[cc] + ghr);
      float zg = sigf(gr[Hdim + cc] + ghz);
      float ng = tanhf(gr[2 * Hdim + cc] + rg * ghn);
      float h = (1.f - zg) * ng + zg * hreg[j];
      hreg[j] = h;
      hout[(size_t)r * Hdim + cc] = (__bf16)h;
      ys[((size_t)t * Bdim + r) * Hdim + cc] = (__bf16)h;
    }
    grid_barrier(bar, (unsigned)REC_BLOCKS * (t + 2));
  }
}

// ---------------------------------------------------------------------------
// Reparameterization: style/content to d_out + bf16 concat for fc GEMM.
__global__ void reparam_kernel(const float* __restrict__ mu_s, const float* __restrict__ lv_s,
                               const float* __restrict__ eps_s, const float* __restrict__ mu_c,
                               const float* __restrict__ lv_c, const float* __restrict__ eps_c,
                               float* __restrict__ o_style, float* __restrict__ o_content,
                               __bf16* __restrict__ sc) {
  int i = blockIdx.x * blockDim.x + threadIdx.x;
  const int NS = 2 * Bdim * Sdim;          // 4096
  const int NC = 2 * Bdim * Cdim;          // 12288
  if (i < NS) {
    float v = mu_s[i] + eps_s[i] * __expf(0.5f * lv_s[i]);
    o_style[i] = v;
    sc[(size_t)(i >> 6) * (Sdim + Cdim) + (i & 63)] = (__bf16)v;
  } else if (i < NS + NC) {
    int k = i - NS;
    float v = mu_c[k] + eps_c[k] * __expf(0.5f * lv_c[k]);
    o_content[k] = v;
    sc[(size_t)(k / Cdim) * (Sdim + Cdim) + Sdim + (k % Cdim)] = (__bf16)v;
  }
}

// ---------------------------------------------------------------------------
extern "C" void kernel_launch(void* const* d_in, const int* in_sizes, int n_in,
                              void* d_out, int out_size, void* d_ws, size_t ws_size,
                              hipStream_t stream) {
  (void)in_sizes; (void)n_in; (void)out_size; (void)ws_size;
  const int*   x      = (const int*)  d_in[0];
  const float* emb    = (const float*)d_in[1];
  const float* ln_g   = (const float*)d_in[2];
  const float* ln_b   = (const float*)d_in[3];
  const float* lw0i   = (const float*)d_in[4];
  const float* lw0h   = (const float*)d_in[5];
  const float* lb0i   = (const float*)d_in[6];
  const float* lb0h   = (const float*)d_in[7];
  const float* lw1i   = (const float*)d_in[8];
  const float* lw1h   = (const float*)d_in[9];
  const float* lb1i   = (const float*)d_in[10];
  const float* lb1h   = (const float*)d_in[11];
  const float* wms    = (const float*)d_in[12];
  const float* bms    = (const float*)d_in[13];
  const float* wvs    = (const float*)d_in[14];
  const float* bvs    = (const float*)d_in[15];
  const float* wmc    = (const float*)d_in[16];
  const float* bmc    = (const float*)d_in[17];
  const float* wvc    = (const float*)d_in[18];
  const float* bvc    = (const float*)d_in[19];
  const float* fc_w   = (const float*)d_in[20];
  const float* fc_b   = (const float*)d_in[21];
  const float* gw0i   = (const float*)d_in[22];
  const float* gw0h   = (const float*)d_in[23];
  const float* gb0i   = (const float*)d_in[24];
  const float* gb0h   = (const float*)d_in[25];
  const float* gw1i   = (const float*)d_in[26];
  const float* gw1h   = (const float*)d_in[27];
  const float* gb1i   = (const float*)d_in[28];
  const float* gb1h   = (const float*)d_in[29];
  const float* out_w  = (const float*)d_in[30];
  const float* out_b  = (const float*)d_in[31];
  const float* eps_s  = (const float*)d_in[32];
  const float* eps_c  = (const float*)d_in[33];
  const int*   sos    = (const int*)  d_in[34];

  float* out       = (float*)d_out;
  float* rec       = out;
  float* o_style   = out + (size_t)Bdim * Tdim * Vdim;
  float* o_content = o_style + 2 * Bdim * Sdim;
  float* o_mus     = o_content + 2 * Bdim * Cdim;
  float* o_lvs     = o_mus + 2 * Bdim * Sdim;
  float* o_muc     = o_lvs + 2 * Bdim * Sdim;
  float* o_lvc     = o_muc + 2 * Bdim * Cdim;

  // ---- workspace carve-out ----
  char* p = (char*)d_ws;
  auto alloc = [&](size_t bytes) -> void* {
    void* r = (void*)p; p += (bytes + 255) & ~(size_t)255; return r;
  };
  unsigned* bar   = (unsigned*)alloc(16 * sizeof(unsigned));
  __bf16* xt      = (__bf16*)alloc((size_t)Tdim * Bdim * Edim * 2);
  __bf16* decin   = (__bf16*)alloc((size_t)Tdim * Bdim * Edim * 2);
  float*  gi      = (float*) alloc((size_t)Tdim * Bdim * 4 * Hdim * 4);  // reused 4x
  __bf16* ys0     = (__bf16*)alloc((size_t)Tdim * Bdim * Hdim * 2);
  __bf16* d0b     = (__bf16*)alloc((size_t)Tdim * Bdim * Hdim * 2);
  __bf16* d1b     = (__bf16*)alloc((size_t)Tdim * Bdim * Hdim * 2);
  __bf16* hbuf    = (__bf16*)alloc((size_t)2 * Bdim * Hdim * 2);
  float*  hfinal  = (float*) alloc((size_t)2 * Bdim * Hdim * 4);
  __bf16* hnb     = (__bf16*)alloc((size_t)2 * Bdim * Hdim * 2);
  float*  zbuf    = (float*) alloc((size_t)2 * Bdim * Hdim * 4);
  __bf16* scb     = (__bf16*)alloc((size_t)2 * Bdim * (Sdim + Cdim) * 2);
  float*  biasL0  = (float*) alloc(4 * Hdim * 4);
  float*  biasL1  = (float*) alloc(4 * Hdim * 4);
  __bf16* bw0i = (__bf16*)alloc((size_t)4 * Hdim * Edim * 2);
  __bf16* bw0h = (__bf16*)alloc((size_t)4 * Hdim * Hdim * 2);
  __bf16* bw1i = (__bf16*)alloc((size_t)4 * Hdim * Hdim * 2);
  __bf16* bw1h = (__bf16*)alloc((size_t)4 * Hdim * Hdim * 2);
  __bf16* bg0i = (__bf16*)alloc((size_t)3 * Hdim * Edim * 2);
  __bf16* bg0h = (__bf16*)alloc((size_t)3 * Hdim * Hdim * 2);
  __bf16* bg1i = (__bf16*)alloc((size_t)3 * Hdim * Hdim * 2);
  __bf16* bg1h = (__bf16*)alloc((size_t)3 * Hdim * Hdim * 2);
  __bf16* bow  = (__bf16*)alloc((size_t)Vdim * Hdim * 2);
  __bf16* bfc  = (__bf16*)alloc((size_t)Hdim * (Sdim + Cdim) * 2);
  __bf16* bwms = (__bf16*)alloc((size_t)Sdim * Hdim * 2);
  __bf16* bwvs = (__bf16*)alloc((size_t)Sdim * Hdim * 2);
  __bf16* bwmc = (__bf16*)alloc((size_t)Cdim * Hdim * 2);
  __bf16* bwvc = (__bf16*)alloc((size_t)Cdim * Hdim * 2);

  hipMemsetAsync(bar, 0, 16 * sizeof(unsigned), stream);

  auto cvt = [&](const float* s, __bf16* d, size_t n) {
    int blocks = (int)((n + 255) / 256); if (blocks > 8192) blocks = 8192;
    cvt_kernel<<<blocks, 256, 0, stream>>>(s, d, n);
  };
  cvt(lw0i, bw0i, (size_t)4 * Hdim * Edim);
  cvt(lw0h, bw0h, (size_t)4 * Hdim * Hdim);
  cvt(lw1i, bw1i, (size_t)4 * Hdim * Hdim);
  cvt(lw1h, bw1h, (size_t)4 * Hdim * Hdim);
  cvt(gw0i, bg0i, (size_t)3 * Hdim * Edim);
  cvt(gw0h, bg0h, (size_t)3 * Hdim * Hdim);
  cvt(gw1i, bg1i, (size_t)3 * Hdim * Hdim);
  cvt(gw1h, bg1h, (size_t)3 * Hdim * Hdim);
  cvt(out_w, bow, (size_t)Vdim * Hdim);
  cvt(fc_w,  bfc, (size_t)Hdim * (Sdim + Cdim));
  cvt(wms, bwms, (size_t)Sdim * Hdim);
  cvt(wvs, bwvs, (size_t)Sdim * Hdim);
  cvt(wmc, bwmc, (size_t)Cdim * Hdim);
  cvt(wvc, bwvc, (size_t)Cdim * Hdim);

  add2_kernel<<<16, 256, 0, stream>>>(lb0i, lb0h, biasL0, 4 * Hdim);
  add2_kernel<<<16, 256, 0, stream>>>(lb1i, lb1h, biasL1, 4 * Hdim);

  embed_ln_kernel<<<Bdim * Tdim, 128, 0, stream>>>(x, emb, ln_g, ln_b, xt, decin);
  sos_kernel<<<(Bdim * Edim + 255) / 256, 256, 0, stream>>>(emb, sos, decin);

  // small GEMM (M=64 heads / fc)
  auto gemm_small = [&](const __bf16* A, const __bf16* W, const float* bias, float* C,
                        int M, int N, int K) {
    int nw = (M >> 4) * (N >> 6);
    gemm_kernel<<<(nw + 7) / 8, 256, 0, stream>>>(A, W, bias, C, M, N, K);
  };
  // blocked GEMM with async-LDS staging (M,N multiples of 128, K of 64)
  auto gemm_big = [&](const __bf16* A, const __bf16* W, const float* bias, float* C,
                      int M, int N, int K, int permute) {
    int blocks = (M >> 7) * (N >> 7);
    gemm_lds_kernel<<<blocks, 256, 0, stream>>>(A, W, bias, C, M, N, K, permute);
  };

  const int MT = Tdim * Bdim;  // 4096

  // LSTM layer 0
  gemm_big(xt, bw0i, biasL0, gi, MT, 4 * Hdim, Edim, 0);
  lstm_rec_kernel<<<REC_BLOCKS, 256, 0, stream>>>(gi, bw0h, hbuf, ys0, hfinal, bar + 0);
  // LSTM layer 1 (ys not needed, only final hidden)
  gemm_big(ys0, bw1i, biasL1, gi, MT, 4 * Hdim, Hdim, 0);
  lstm_rec_kernel<<<REC_BLOCKS, 256, 0, stream>>>(gi, bw1h, hbuf, ((__bf16*)0),
                                                  hfinal + (size_t)Bdim * Hdim, bar + 1);

  // VAE heads on hn = [h0; h1]  (bf16 for WMMA A)
  cvt(hfinal, hnb, (size_t)2 * Bdim * Hdim);
  gemm_small(hnb, bwms, bms, o_mus, 2 * Bdim, Sdim, Hdim);
  gemm_small(hnb, bwvs, bvs, o_lvs, 2 * Bdim, Sdim, Hdim);
  gemm_small(hnb, bwmc, bmc, o_muc, 2 * Bdim, Cdim, Hdim);
  gemm_small(hnb, bwvc, bvc, o_lvc, 2 * Bdim, Cdim, Hdim);
  reparam_kernel<<<(2 * Bdim * (Sdim + Cdim) + 255) / 256, 256, 0, stream>>>(
      o_mus, o_lvs, eps_s, o_muc, o_lvc, eps_c, o_style, o_content, scb);
  gemm_small(scb, bfc, fc_b, zbuf, 2 * Bdim, Hdim, Sdim + Cdim);

  // GRU decoder layer 0 (h0 = z[0])
  gemm_big(decin, bg0i, gb0i, gi, MT, 3 * Hdim, Edim, 0);
  gru_rec_kernel<<<REC_BLOCKS, 256, 0, stream>>>(gi, bg0h, gb0h, zbuf, hbuf, d0b, bar + 2);
  // GRU decoder layer 1 (h0 = z[1])
  gemm_big(d0b, bg1i, gb1i, gi, MT, 3 * Hdim, Hdim, 0);
  gru_rec_kernel<<<REC_BLOCKS, 256, 0, stream>>>(gi, bg1h, gb1h,
                                                 zbuf + (size_t)Bdim * Hdim, hbuf, d1b, bar + 3);

  // Vocab projection: blocked GEMM + fused (t,b)->(b,t) row permutation
  gemm_big(d1b, bow, out_b, rec, MT, Vdim, Hdim, 1);
}
